// BERT_36146444763559
// MI455X (gfx1250) — matrix operand order repaired and checked by reading, not compiled, not run on hardware
//
#include <hip/hip_runtime.h>
#include <math.h>

// ---- model constants (from reference) ----
#define Vv   32000
#define Dd   1024
#define Hh   16
#define Ll   8
#define Tt   512
#define HSs  64
#define Bb   8
#define BTt  4096   // B*T
#define FFd  4096   // 4*D

typedef __bf16 bf16;
typedef __bf16 v16bf  __attribute__((ext_vector_type(16)));
typedef __bf16 bf16x8 __attribute__((ext_vector_type(8)));
typedef float  v8f    __attribute__((ext_vector_type(8)));

// ---------------------------------------------------------------------------
// WMMA helpers: CDNA5 v_wmma_f32_16x16x32_bf16 fragment loaders.
// A (16x32, 16-bit): lane holds row = lane&15; K chunks [8*(lane>>4) .. +7]
//                    and [8*(lane>>4)+16 .. +23]  (two 16B LDS loads).
// B (32x16, 16-bit): lane holds col = lane&15; K chunks [8*(lane>>4) .. +7]
//                    and [8*(lane>>4)+16 .. +23] when B is staged transposed as Bt[N][K].
// ---------------------------------------------------------------------------
__device__ __forceinline__ v16bf frag_a(const bf16* base, int ld) {
  const int lane = threadIdx.x & 31;
  const int row = lane & 15, hi = lane >> 4;
  union { v16bf v; bf16x8 h[2]; } u;
  u.h[0] = *(const bf16x8*)(base + row * ld + hi * 8);
  u.h[1] = *(const bf16x8*)(base + row * ld + hi * 8 + 16);
  return u.v;
}
__device__ __forceinline__ v16bf frag_b(const bf16* base, int ld) {
  const int lane = threadIdx.x & 31;
  const int col = lane & 15, hi = lane >> 4;
  union { v16bf v; bf16x8 h[2]; } u;
  u.h[0] = *(const bf16x8*)(base + col * ld + hi * 8);
  u.h[1] = *(const bf16x8*)(base + col * ld + hi * 8 + 16);
  return u.v;
}
__device__ __forceinline__ v8f wmma_bf16(v16bf a, v16bf b, v8f c) {
  return __builtin_amdgcn_wmma_f32_16x16x32_bf16(false, a, false, b,
                                                 (short)0, c, false, false);
}

// ---------------------------------------------------------------------------
// Weight conversion kernels (f32 -> bf16, run once per launch)
// ---------------------------------------------------------------------------
__global__ void cvt_elem_kernel(const float* __restrict__ s,
                                bf16* __restrict__ d, long n) {
  long i = (long)blockIdx.x * blockDim.x + threadIdx.x;
  if (i < n) d[i] = (bf16)s[i];
}

// (L,H,D,HS) -> (L, D, H*HS): dst[(l*D + dRow)*D + h*HS+e] = src[((l*H+h)*D+dRow)*HS+e]
__global__ void cvt_qkv_kernel(const float* __restrict__ s,
                               bf16* __restrict__ d) {
  long i = (long)blockIdx.x * blockDim.x + threadIdx.x;
  const long n = (long)Ll * Dd * Dd;
  if (i >= n) return;
  int l  = (int)(i >> 20);            // / (D*D)
  int r  = (int)(i & ((1 << 20) - 1));
  int dR = r >> 10;                   // / D
  int he = r & 1023;
  int h  = he >> 6;
  int e  = he & 63;
  d[i] = (bf16)s[(((long)l * Hh + h) * Dd + dR) * HSs + e];
}

// ---------------------------------------------------------------------------
// Embedding: x[b,t,:] = tok[idx[b,t],:] + pos[t,:]
// ---------------------------------------------------------------------------
__global__ void embed_kernel(const int* __restrict__ idx,
                             const float* __restrict__ tok,
                             const float* __restrict__ pos,
                             float* __restrict__ x) {
  long i = (long)blockIdx.x * blockDim.x + threadIdx.x;
  if (i >= (long)BTt * Dd) return;
  int token = (int)(i >> 10);
  int dcol  = (int)(i & 1023);
  int tpos  = token & (Tt - 1);
  x[i] = tok[(long)idx[token] * Dd + dcol] + pos[(long)tpos * Dd + dcol];
}

// ---------------------------------------------------------------------------
// RMSNorm over D=1024, one block per row, emits bf16 for the next GEMM
// ---------------------------------------------------------------------------
__global__ __launch_bounds__(256)
void rmsnorm_kernel(const float* __restrict__ x, const float* __restrict__ w,
                    bf16* __restrict__ out) {
  const int row = blockIdx.x;
  const int t = threadIdx.x;
  const float* xr = x + (long)row * Dd;
  __shared__ float red[256];
  __shared__ float rs;
  float v[4];
  float s = 0.f;
#pragma unroll
  for (int j = 0; j < 4; ++j) {
    v[j] = xr[t + j * 256];
    s += v[j] * v[j];
  }
  red[t] = s;
  __syncthreads();
  for (int k = 128; k > 0; k >>= 1) {
    if (t < k) red[t] += red[t + k];
    __syncthreads();
  }
  if (t == 0) rs = rsqrtf(red[0] * (1.0f / Dd) + 1e-6f);
  __syncthreads();
  const float r = rs;
#pragma unroll
  for (int j = 0; j < 4; ++j) {
    int i = t + j * 256;
    out[(long)row * Dd + i] = (bf16)(v[j] * r * w[i]);
  }
}

// ---------------------------------------------------------------------------
// GEMM: C[M,N] = act( A_bf16[M,K] * W_bf16[K,N] + bias [+ residual] )
// block tile 128x128, 8 waves, wave tile 64x32, K-step 32.
// Output config is compile-time (no per-element branches in the epilogue).
// ---------------------------------------------------------------------------
#define BM 128
#define BN 128
#define BK 32
#define LDT 40  // padded K stride in LDS (conflict-free b128 frag loads)

template <bool RES, bool LEAKY, bool OUTF, bool OUTB>
__global__ __launch_bounds__(256)
void gemm_bf16_kernel(const bf16* __restrict__ A, const bf16* __restrict__ W,
                      const float* __restrict__ bias,
                      const float* __restrict__ res,
                      float* __restrict__ outF, bf16* __restrict__ outB,
                      int M, int N, int K) {
  __shared__ bf16 aT[BM * LDT];
  __shared__ bf16 bT[BN * LDT];
  const int t = threadIdx.x;
  const int wid = t >> 5;
  const int m0 = blockIdx.y * BM;
  const int n0 = blockIdx.x * BN;
  const int wm = (wid >> 2) * 64;  // 0 | 64
  const int wn = (wid & 3) * 32;   // 0..96

  v8f acc[4][2];
#pragma unroll
  for (int mt = 0; mt < 4; ++mt)
#pragma unroll
    for (int nt = 0; nt < 2; ++nt)
      acc[mt][nt] = (v8f){0.f, 0.f, 0.f, 0.f, 0.f, 0.f, 0.f, 0.f};

  for (int k0 = 0; k0 < K; k0 += BK) {
    __syncthreads();
    // stage A: 128 rows x 32 k, 16B coalesced chunks (512 / 256 threads)
#pragma unroll
    for (int i = 0; i < 2; ++i) {
      int cid = t + i * 256;
      int r = cid >> 2, ko = (cid & 3) * 8;
      *(bf16x8*)&aT[r * LDT + ko] =
          *(const bf16x8*)&A[(long)(m0 + r) * K + k0 + ko];
    }
    // stage B transposed: coalesced 16B global reads, scalar LDS scatter
#pragma unroll
    for (int i = 0; i < 2; ++i) {
      int cid = t + i * 256;            // 512 chunks of 8
      int kk = cid >> 4;                // 32 k-rows, 16 chunks each
      int nn = (cid & 15) * 8;
      bf16x8 w8 = *(const bf16x8*)&W[(long)(k0 + kk) * N + n0 + nn];
#pragma unroll
      for (int j = 0; j < 8; ++j) bT[(nn + j) * LDT + kk] = w8[j];
    }
    // prefetch next K-tile into cache while WMMAs run (global_prefetch_b8)
    if (k0 + BK < K) {
      __builtin_prefetch(&A[(long)(m0 + (t & 127)) * K + k0 + BK], 0, 0);
      __builtin_prefetch(
          &W[(long)(k0 + BK + (t & 31)) * N + n0 + ((t >> 5) & 3) * 32], 0, 0);
    }
    __syncthreads();

    v16bf bf0 = frag_b(&bT[(wn + 0) * LDT], LDT);
    v16bf bf1 = frag_b(&bT[(wn + 16) * LDT], LDT);
#pragma unroll
    for (int mt = 0; mt < 4; ++mt) {
      v16bf af = frag_a(&aT[(wm + mt * 16) * LDT], LDT);
      acc[mt][0] = wmma_bf16(af, bf0, acc[mt][0]);
      acc[mt][1] = wmma_bf16(af, bf1, acc[mt][1]);
    }
  }

  // epilogue: C-fragment layout -> row = r + 8*(lane>>4), col = lane&15
  const int lane = t & 31, hi = lane >> 4, lo = lane & 15;
  float bv[2];
#pragma unroll
  for (int nt = 0; nt < 2; ++nt) bv[nt] = bias[n0 + wn + nt * 16 + lo];
#pragma unroll
  for (int mt = 0; mt < 4; ++mt) {
#pragma unroll
    for (int nt = 0; nt < 2; ++nt) {
      const long rowb = (long)(m0 + wm + mt * 16 + 8 * hi);
      const int col = n0 + wn + nt * 16 + lo;
#pragma unroll
      for (int r = 0; r < 8; ++r) {
        const long o = (rowb + r) * N + col;
        float v = acc[mt][nt][r] + bv[nt];
        if constexpr (RES) v += res[o];
        if constexpr (LEAKY) v = v >= 0.f ? v : 0.1f * v;
        if constexpr (OUTF) outF[o] = v;
        if constexpr (OUTB) outB[o] = (bf16)v;
      }
    }
  }
}

// ---------------------------------------------------------------------------
// Flash attention: one block = (b, h, 64-row q tile); 4 waves, 16 q rows each.
// Q,K,V layout: [(b*T + t)*D + h*HS + e] (bf16).  scale = sqrt(HS) = 8
// (faithful to reference's `scores * HS**0.5`).
// ---------------------------------------------------------------------------
#define FLD 72  // padded inner stride

__global__ __launch_bounds__(128)
void flash_attn_kernel(const bf16* __restrict__ Q, const bf16* __restrict__ Kb,
                       const bf16* __restrict__ Vb, bf16* __restrict__ O,
                       int masked) {
  const int qt = blockIdx.x, h = blockIdx.y, b = blockIdx.z;
  const int q0 = qt * 64;
  __shared__ bf16 qs[64 * FLD];   // Q tile [qrow][e]
  __shared__ bf16 ks[64 * FLD];   // K tile [key][e]   (== Bt[N=key][K=e])
  __shared__ bf16 vs[64 * FLD];   // V tile transposed [e][key]
  __shared__ float ss[64 * FLD];  // scores f32
  __shared__ bf16 ps[64 * FLD];   // probs bf16 (A-matrix for P*V)
  __shared__ float alpha[64];

  const int t = threadIdx.x;
  const int wid = t >> 5, lane = t & 31, lo = lane & 15, hi = lane >> 4;

  // stage Q once
#pragma unroll
  for (int i = 0; i < 4; ++i) {
    int cid = t + i * 128;           // 512 chunks of 8 bf16
    int r = cid >> 3, eo = (cid & 7) * 8;
    *(bf16x8*)&qs[r * FLD + eo] =
        *(const bf16x8*)&Q[((long)(b * Tt + q0 + r)) * Dd + h * HSs + eo];
  }

  v8f o[4];
#pragma unroll
  for (int nt = 0; nt < 4; ++nt)
    o[nt] = (v8f){0.f, 0.f, 0.f, 0.f, 0.f, 0.f, 0.f, 0.f};
  float m_r = -INFINITY, l_r = 0.f;  // per-row stats (lanes 0..15)

  const int nkt = masked ? (qt + 1) : (Tt / 64);
  for (int kt = 0; kt < nkt; ++kt) {
    const int kb0 = kt * 64;
    __syncthreads();  // protect ks/vs/ps reuse from previous iteration
    // stage K tile [key][e], coalesced 16B chunks
#pragma unroll
    for (int i = 0; i < 4; ++i) {
      int cid = t + i * 128;
      int r = cid >> 3, eo = (cid & 7) * 8;
      *(bf16x8*)&ks[r * FLD + eo] =
          *(const bf16x8*)&Kb[((long)(b * Tt + kb0 + r)) * Dd + h * HSs + eo];
    }
    // stage V transposed [e][key]: coalesced 16B reads, scalar LDS scatter
#pragma unroll
    for (int i = 0; i < 4; ++i) {
      int cid = t + i * 128;          // 512 chunks of 8
      int r = cid >> 3, e0 = (cid & 7) * 8;  // r = key
      bf16x8 v8 =
          *(const bf16x8*)&Vb[((long)(b * Tt + kb0 + r)) * Dd + h * HSs + e0];
#pragma unroll
      for (int j = 0; j < 8; ++j) vs[(e0 + j) * FLD + r] = v8[j];
    }
    __syncthreads();

    // S = Q * K^T  (wave rows [wid*16 .. +15], 64 keys)
#pragma unroll
    for (int nt = 0; nt < 4; ++nt) {
      v8f s = (v8f){0.f, 0.f, 0.f, 0.f, 0.f, 0.f, 0.f, 0.f};
#pragma unroll
      for (int kk = 0; kk < 64; kk += 32) {
        v16bf af = frag_a(&qs[(wid * 16) * FLD + kk], FLD);
        v16bf bfv = frag_b(&ks[(nt * 16) * FLD + kk], FLD);
        s = wmma_bf16(af, bfv, s);
      }
#pragma unroll
      for (int r = 0; r < 8; ++r) {
        int lrow = wid * 16 + r + 8 * hi;
        int rg = q0 + lrow;
        int cg = kb0 + nt * 16 + lo;
        float v = s[r] * 8.0f;  // * HS**0.5
        if (masked && cg > rg) v = -INFINITY;
        ss[lrow * FLD + nt * 16 + lo] = v;
      }
    }
    __syncthreads();

    // online softmax: lanes 0..15 each own one row of this wave's 16 rows
    if (lane < 16) {
      int r = wid * 16 + lane;
      float mx = m_r;
#pragma unroll 8
      for (int j = 0; j < 64; ++j) mx = fmaxf(mx, ss[r * FLD + j]);
      float corr = __expf(m_r - mx);
      float sum = 0.f;
#pragma unroll 8
      for (int j = 0; j < 64; ++j) {
        float p = __expf(ss[r * FLD + j] - mx);
        sum += p;
        ps[r * FLD + j] = (bf16)p;
      }
      l_r = l_r * corr + sum;
      m_r = mx;
      alpha[r] = corr;
    }
    __syncthreads();

    // rescale O accumulators by alpha[row]
    {
      const float4 a0 = *(const float4*)&alpha[wid * 16 + 8 * hi];
      const float4 a1 = *(const float4*)&alpha[wid * 16 + 8 * hi + 4];
      const float ar[8] = {a0.x, a0.y, a0.z, a0.w, a1.x, a1.y, a1.z, a1.w};
#pragma unroll
      for (int nt = 0; nt < 4; ++nt)
#pragma unroll
        for (int r = 0; r < 8; ++r) o[nt][r] *= ar[r];
    }

    // O += P * V
#pragma unroll
    for (int nt = 0; nt < 4; ++nt) {
#pragma unroll
      for (int kk = 0; kk < 64; kk += 32) {
        v16bf af = frag_a(&ps[(wid * 16) * FLD + kk], FLD);
        v16bf bfv = frag_b(&vs[(nt * 16) * FLD + kk], FLD);
        o[nt] = wmma_bf16(af, bfv, o[nt]);
      }
    }
  }

  // finalize: O *= 1/l, store bf16 in (b,t,h,e) layout
  if (lane < 16) alpha[wid * 16 + lane] = 1.0f / l_r;
  __syncthreads();
  {
    const float4 a0 = *(const float4*)&alpha[wid * 16 + 8 * hi];
    const float4 a1 = *(const float4*)&alpha[wid * 16 + 8 * hi + 4];
    const float ar[8] = {a0.x, a0.y, a0.z, a0.w, a1.x, a1.y, a1.z, a1.w};
#pragma unroll
    for (int nt = 0; nt < 4; ++nt) {
#pragma unroll
      for (int r = 0; r < 8; ++r) {
        int trow = q0 + wid * 16 + r + 8 * hi;
        int e = nt * 16 + lo;
        O[((long)(b * Tt + trow)) * Dd + h * HSs + e] = (bf16)(o[nt][r] * ar[r]);
      }
    }
  }
}

// ---------------------------------------------------------------------------
// host side
// ---------------------------------------------------------------------------
static void launch_gemm(const bf16* A, const bf16* W, const float* bias,
                        const float* res, float* outF, bf16* outB, int M,
                        int N, int K, int leaky, hipStream_t s) {
  dim3 g(N / BN, M / BM);
  if (res) {        // residual add, f32 output (attn proj / FFN2)
    gemm_bf16_kernel<true, false, true, false>
        <<<g, 256, 0, s>>>(A, W, bias, res, outF, outB, M, N, K);
  } else if (leaky) {  // leaky-ReLU, bf16 output (FFN1)
    gemm_bf16_kernel<false, true, false, true>
        <<<g, 256, 0, s>>>(A, W, bias, res, outF, outB, M, N, K);
  } else if (outF) {   // plain f32 output (logits)
    gemm_bf16_kernel<false, false, true, false>
        <<<g, 256, 0, s>>>(A, W, bias, res, outF, outB, M, N, K);
  } else {             // plain bf16 output (Q/K/V)
    gemm_bf16_kernel<false, false, false, true>
        <<<g, 256, 0, s>>>(A, W, bias, res, outF, outB, M, N, K);
  }
}

extern "C" void kernel_launch(void* const* d_in, const int* in_sizes, int n_in,
                              void* d_out, int out_size, void* d_ws,
                              size_t ws_size, hipStream_t stream) {
  (void)in_sizes; (void)n_in; (void)out_size; (void)ws_size;
  const int*   idx   = (const int*)d_in[0];
  const float* tok   = (const float*)d_in[1];
  const float* pos   = (const float*)d_in[2];
  const float* Wq    = (const float*)d_in[3];
  const float* bq    = (const float*)d_in[4];
  const float* Wk    = (const float*)d_in[5];
  const float* bk    = (const float*)d_in[6];
  const float* Wv    = (const float*)d_in[7];
  const float* bv    = (const float*)d_in[8];
  const float* Wo    = (const float*)d_in[9];
  const float* bo    = (const float*)d_in[10];
  const float* W1    = (const float*)d_in[11];
  const float* b1    = (const float*)d_in[12];
  const float* W2    = (const float*)d_in[13];
  const float* b2    = (const float*)d_in[14];
  const float* nw    = (const float*)d_in[15];
  const float* fnw   = (const float*)d_in[16];
  const float* Wf    = (const float*)d_in[17];
  const float* bf    = (const float*)d_in[18];
  float* logits = (float*)d_out;

  // workspace bump allocator (256B aligned)
  char* ws = (char*)d_ws;
  size_t off = 0;
  auto alloc = [&](size_t bytes) -> char* {
    char* p = ws + off;
    off += (bytes + 255) & ~(size_t)255;
    return p;
  };
  const size_t LDD = (size_t)Ll * Dd * Dd;  // 8M elems
  bf16* wq_b = (bf16*)alloc(LDD * 2);
  bf16* wk_b = (bf16*)alloc(LDD * 2);
  bf16* wv_b = (bf16*)alloc(LDD * 2);
  bf16* wo_b = (bf16*)alloc(LDD * 2);
  bf16* w1_b = (bf16*)alloc((size_t)Ll * Dd * FFd * 2);
  bf16* w2_b = (bf16*)alloc((size_t)Ll * FFd * Dd * 2);
  bf16* wf_b = (bf16*)alloc((size_t)Dd * Vv * 2);
  float* X   = (float*)alloc((size_t)BTt * Dd * 4);
  float* X2  = (float*)alloc((size_t)BTt * Dd * 4);
  bf16* XN   = (bf16*)alloc((size_t)BTt * Dd * 2);
  bf16* Qb   = (bf16*)alloc((size_t)BTt * Dd * 2);
  bf16* Kbuf = (bf16*)alloc((size_t)BTt * Dd * 2);
  bf16* Vbuf = (bf16*)alloc((size_t)BTt * Dd * 2);
  bf16* AO   = (bf16*)alloc((size_t)BTt * Dd * 2);
  bf16* Hb   = (bf16*)alloc((size_t)BTt * FFd * 2);

  // one-time weight conversion (f32 -> bf16)
  {
    long n = (long)LDD;
    int blk = (int)((n + 255) / 256);
    cvt_qkv_kernel<<<blk, 256, 0, stream>>>(Wq, wq_b);
    cvt_qkv_kernel<<<blk, 256, 0, stream>>>(Wk, wk_b);
    cvt_qkv_kernel<<<blk, 256, 0, stream>>>(Wv, wv_b);
    cvt_elem_kernel<<<blk, 256, 0, stream>>>(Wo, wo_b, n);
    long n14 = (long)Ll * Dd * FFd;
    cvt_elem_kernel<<<(int)((n14 + 255) / 256), 256, 0, stream>>>(W1, w1_b, n14);
    cvt_elem_kernel<<<(int)((n14 + 255) / 256), 256, 0, stream>>>(W2, w2_b, n14);
    long nf = (long)Dd * Vv;
    cvt_elem_kernel<<<(int)((nf + 255) / 256), 256, 0, stream>>>(Wf, wf_b, nf);
  }

  // embedding
  embed_kernel<<<(BTt * Dd) / 256, 256, 0, stream>>>(idx, tok, pos, X);

  const dim3 fgrid(Tt / 64, Hh, Bb);
  for (int l = 0; l < Ll; ++l) {
    const bf16* wql = wq_b + (size_t)l * Dd * Dd;
    const bf16* wkl = wk_b + (size_t)l * Dd * Dd;
    const bf16* wvl = wv_b + (size_t)l * Dd * Dd;
    const bf16* wol = wo_b + (size_t)l * Dd * Dd;
    const bf16* w1l = w1_b + (size_t)l * Dd * FFd;
    const bf16* w2l = w2_b + (size_t)l * FFd * Dd;
    const float* bql = bq + (size_t)l * Dd;
    const float* bkl = bk + (size_t)l * Dd;
    const float* bvl = bv + (size_t)l * Dd;
    const float* bol = bo + (size_t)l * Dd;
    const float* b1l = b1 + (size_t)l * FFd;
    const float* b2l = b2 + (size_t)l * Dd;
    const float* nwl = nw + (size_t)l * Dd;

    // ---- unmasked attention, residual from X -> X2 ----
    rmsnorm_kernel<<<BTt, 256, 0, stream>>>(X, nwl, XN);
    launch_gemm(XN, wql, bql, nullptr, nullptr, Qb, BTt, Dd, Dd, 0, stream);
    launch_gemm(XN, wkl, bkl, nullptr, nullptr, Kbuf, BTt, Dd, Dd, 0, stream);
    launch_gemm(XN, wvl, bvl, nullptr, nullptr, Vbuf, BTt, Dd, Dd, 0, stream);
    flash_attn_kernel<<<fgrid, 128, 0, stream>>>(Qb, Kbuf, Vbuf, AO, 0);
    launch_gemm(AO, wol, bol, X, X2, nullptr, BTt, Dd, Dd, 0, stream);

    // ---- FFN (residual from ORIGINAL X) -> X2 ----
    rmsnorm_kernel<<<BTt, 256, 0, stream>>>(X2, nwl, XN);
    launch_gemm(XN, w1l, b1l, nullptr, nullptr, Hb, BTt, FFd, Dd, 1, stream);
    launch_gemm(Hb, w2l, b2l, X, X2, nullptr, BTt, Dd, FFd, 0, stream);

    // ---- masked attention (same weights), residual from X2 -> X ----
    rmsnorm_kernel<<<BTt, 256, 0, stream>>>(X2, nwl, XN);
    launch_gemm(XN, wql, bql, nullptr, nullptr, Qb, BTt, Dd, Dd, 0, stream);
    launch_gemm(XN, wkl, bkl, nullptr, nullptr, Kbuf, BTt, Dd, Dd, 0, stream);
    launch_gemm(XN, wvl, bvl, nullptr, nullptr, Vbuf, BTt, Dd, Dd, 0, stream);
    flash_attn_kernel<<<fgrid, 128, 0, stream>>>(Qb, Kbuf, Vbuf, AO, 1);
    launch_gemm(AO, wol, bol, X2, X, nullptr, BTt, Dd, Dd, 0, stream);
  }

  // final norm + logits
  rmsnorm_kernel<<<BTt, 256, 0, stream>>>(X, fnw, XN);
  launch_gemm(XN, wf_b, bf, nullptr, logits, nullptr, BTt, Vv, Dd, 0, stream);
}
